// Conv2d_STN_15762529976344
// MI455X (gfx1250) — matrix operand order repaired and checked by reading, not compiled, not run on hardware
//
#include <hip/hip_runtime.h>
#include <hip/hip_bf16.h>

typedef __attribute__((ext_vector_type(16))) _Float16 v16h;
typedef __attribute__((ext_vector_type(8)))  _Float16 v8h;
typedef __attribute__((ext_vector_type(8)))  float    v8f;

// ---- problem constants ----
#define Bsz   8
#define CH    64
#define Hh    96
#define Ww    96
#define HW    9216          // 96*96
#define Ho    94
#define Wo    94
#define P     8836          // 94*94
#define Ntot  70688         // Bsz*P
#define Ktot  576           // CH*9
#define KB    18            // Ktot/32
#define Ff    128
#define NC    20
#define LDA   584           // 576 + 8 halves pad (keeps 16B align, breaks bank stride)

// d_out layout (floats)
#define OFF_LIKE   0
#define OFF_PROBS  160
#define OFF_BOX    (160 + 1413760)
#define OFF_BOXNT  (OFF_BOX + 40)
#define OFF_THETA  (OFF_BOXNT + 40)

// ---------------------------------------------------------------------------
// Pack B matrix (logical [N][K=576], row-major) into WMMA-fragment-contiguous
// f16: bp[((kb*Ntiles+nt)*32+lane)*16 + t] = W[n][k],
//   n = nt*16 + (lane&15), k = kb*32 + (lane>>4)*8 + (t&7) + (t>>3)*16   (B 32x16 f16 layout)
// ---------------------------------------------------------------------------
__global__ void pack_b_kernel(const float* __restrict__ W, _Float16* __restrict__ bp,
                              int Ntiles) {
    int idx = blockIdx.x * 256 + threadIdx.x;
    int total = KB * Ntiles * 32 * 16;
    if (idx >= total) return;
    int t    = idx & 15;
    int lane = (idx >> 4) & 31;
    int knt  = idx >> 9;            // kb*Ntiles + nt
    int nt   = knt % Ntiles;
    int kb   = knt / Ntiles;
    int k = kb * 32 + (lane >> 4) * 8 + (t & 7) + (t >> 3) * 16;
    int n = nt * 16 + (lane & 15);
    bp[idx] = (_Float16)W[(size_t)n * Ktot + k];
}

// ---------------------------------------------------------------------------
// conv0: implicit-GEMM WMMA. One block = 16 output pixels x 64 channels.
// Stage im2col rows (16x576 f16) in LDS, 4 waves each do a 16x16 tile.
// h = relu(conv(x, stn0_w) + stn0_b), stored fp32 [B][C][H][W].
// ---------------------------------------------------------------------------
__global__ __launch_bounds__(128) void conv0_wmma_kernel(
    const float* __restrict__ x, const _Float16* __restrict__ bp0,
    const float* __restrict__ b0, float* __restrict__ hbuf) {
    __shared__ _Float16 lA[16 * LDA];
    int tid   = threadIdx.x;
    int mBase = blockIdx.x * 16;         // 16 pixels, same b and same row y
    int b     = mBase / HW;
    int rem   = mBase % HW;
    int y     = rem / Ww;
    int x0    = rem % Ww;

    for (int i = tid; i < 16 * Ktot; i += 128) {
        int r = i / Ktot, k = i % Ktot;
        int c  = k / 9;
        int kh = (k % 9) / 3;
        int kw = k % 3;
        int yy = y + kh - 1;
        int xx = x0 + r + kw - 1;
        float v = 0.0f;
        if (yy >= 0 && yy < Hh && xx >= 0 && xx < Ww)
            v = x[((size_t)(b * CH + c) * Hh + yy) * Ww + xx];
        lA[r * LDA + k] = (_Float16)v;
    }
    __syncthreads();

    int wave = tid >> 5, lane = tid & 31;
    int m = lane & 15, khalf = lane >> 4;
    v8f acc = {};
    for (int kb = 0; kb < KB; ++kb) {
        const _Float16* ap = &lA[m * LDA + kb * 32 + khalf * 8];
        v8h a0 = *(const v8h*)ap;
        v8h a1 = *(const v8h*)(ap + 16);
        v16h af = __builtin_shufflevector(a0, a1, 0,1,2,3,4,5,6,7,8,9,10,11,12,13,14,15);
        const _Float16* bptr = bp0 + ((size_t)(kb * 4 + wave) * 32 + lane) * 16;
        if (kb + 1 < KB)
            __builtin_prefetch(bp0 + ((size_t)((kb + 1) * 4 + wave) * 32 + lane) * 16, 0, 0);
        v16h bf = *(const v16h*)bptr;
        acc = __builtin_amdgcn_wmma_f32_16x16x32_f16(false, af, false, bf,
                                                     (short)0, acc, false, false);
    }
    int col   = wave * 16 + (lane & 15);
    int rbase = (lane >> 4) * 8;
    float bias = b0[col];
    for (int v = 0; v < 8; ++v) {
        int pix = rem + rbase + v;   // same b
        hbuf[(size_t)(b * CH + col) * HW + pix] = fmaxf(acc[v] + bias, 0.0f);
    }
}

// ---------------------------------------------------------------------------
// conv1 (64->6, no pad) + identity; also emit output5 = identity - theta.
// theta ordering n_t = b*P + wo*Ho + ho, theta[n_t][j] = out6[b][j][ho][wo].
// ---------------------------------------------------------------------------
__global__ void conv1_theta_kernel(const float* __restrict__ hbuf,
                                   const float* __restrict__ w1,
                                   float* __restrict__ out6,
                                   float* __restrict__ theta_out) {
    int t = blockIdx.x * 256 + threadIdx.x;
    if (t >= Ntot) return;
    int b = t / P, s = t % P;
    int ho = s / Wo, wo = s % Wo;
    float acc[6] = {1.f, 0.f, 0.f, 0.f, 1.f, 0.f};
    const float* hb = hbuf + (size_t)b * CH * HW;
    for (int c = 0; c < CH; ++c) {
        const float* hp = hb + c * HW + ho * Ww + wo;
        for (int kh = 0; kh < 3; ++kh)
            for (int kw = 0; kw < 3; ++kw) {
                float hv = hp[kh * Ww + kw];
                for (int j = 0; j < 6; ++j)
                    acc[j] += hv * w1[((size_t)(j * CH + c)) * 9 + kh * 3 + kw];
            }
    }
    const float id6[6] = {1.f, 0.f, 0.f, 0.f, 1.f, 0.f};
    for (int j = 0; j < 6; ++j) {
        out6[((size_t)(b * 6 + j) * Ho + ho) * Wo + wo] = acc[j];
        theta_out[((size_t)(b * P + wo * Ho + ho)) * 6 + j] = id6[j] - acc[j];
    }
}

__device__ __forceinline__ float bilin96(const float* __restrict__ img,
                                         float ix, float iy) {
    float fx = floorf(ix), fy = floorf(iy);
    int x0 = (int)fx, y0 = (int)fy;
    float wx1 = ix - fx, wy1 = iy - fy;
    float wx0 = 1.0f - wx1, wy0 = 1.0f - wy1;
    int x1 = x0 + 1, y1 = y0 + 1;
    float v = 0.0f;
    if (y0 >= 0 && y0 < Hh && x0 >= 0 && x0 < Ww) v += img[y0 * Ww + x0] * (wy0 * wx0);
    if (y0 >= 0 && y0 < Hh && x1 >= 0 && x1 < Ww) v += img[y0 * Ww + x1] * (wy0 * wx1);
    if (y1 >= 0 && y1 < Hh && x0 >= 0 && x0 < Ww) v += img[y1 * Ww + x0] * (wy1 * wx0);
    if (y1 >= 0 && y1 < Hh && x1 >= 0 && x1 < Ww) v += img[y1 * Ww + x1] * (wy1 * wx1);
    return v;
}

// ---------------------------------------------------------------------------
// Fused bilinear-gather + linear GEMM (the dominant 10.4 GFLOP op).
// One block = 16 rows (pixels) x 128 output features, 8 waves.
// Stage: tap coords -> LDS, then 16x576 gathered patch (f16) -> LDS,
// then WMMA over 18 k-blocks with pre-packed W_lin fragments.
// theta at pixel (ho,wo) = out6[b][:, wo, ho] (replicates reference quirk);
// ix = gx*1.5 + ho + 1.5 ; iy = gy*1.5 + wo + 1.5.
// ---------------------------------------------------------------------------
__global__ __launch_bounds__(256) void gather_linear_wmma_kernel(
    const float* __restrict__ x, const float* __restrict__ out6,
    const _Float16* __restrict__ bpl, const float* __restrict__ b_lin,
    float* __restrict__ xout) {
    __shared__ _Float16 lA[16 * LDA];
    __shared__ float cIx[16][9];
    __shared__ float cIy[16][9];
    int tid   = threadIdx.x;
    int nBase = blockIdx.x * 16;

    if (tid < 144) {
        int r = tid / 9, tap = tid % 9;
        int n = nBase + r;
        int b = n / P, s = n % P;
        int ho = s / Wo, wo = s % Wo;
        const float* o6 = out6 + (size_t)b * 6 * P + wo * Wo + ho;
        float t0 = o6[0],     t1 = o6[P],     t2 = o6[2 * P];
        float t3 = o6[3 * P], t4 = o6[4 * P], t5 = o6[5 * P];
        float cxv = (float)(tap % 3) - 1.0f;
        float cyv = (float)(tap / 3) - 1.0f;
        cIx[r][tap] = (t0 * cxv + t1 * cyv + t2) * 1.5f + (float)ho + 1.5f;
        cIy[r][tap] = (t3 * cxv + t4 * cyv + t5) * 1.5f + (float)wo + 1.5f;
    }
    __syncthreads();

    for (int i = tid; i < 16 * Ktot; i += 256) {
        int r = i / Ktot, k = i % Ktot;
        int c = k / 9, tap = k % 9;
        int n = nBase + r;
        int b = n / P;
        const float* img = x + (size_t)(b * CH + c) * HW;
        lA[r * LDA + k] = (_Float16)bilin96(img, cIx[r][tap], cIy[r][tap]);
    }
    __syncthreads();

    int wave = tid >> 5, lane = tid & 31;
    int m = lane & 15, khalf = lane >> 4;
    v8f acc = {};
    for (int kb = 0; kb < KB; ++kb) {
        const _Float16* ap = &lA[m * LDA + kb * 32 + khalf * 8];
        v8h a0 = *(const v8h*)ap;
        v8h a1 = *(const v8h*)(ap + 16);
        v16h af = __builtin_shufflevector(a0, a1, 0,1,2,3,4,5,6,7,8,9,10,11,12,13,14,15);
        const _Float16* bptr = bpl + ((size_t)(kb * 8 + wave) * 32 + lane) * 16;
        if (kb + 1 < KB)
            __builtin_prefetch(bpl + ((size_t)((kb + 1) * 8 + wave) * 32 + lane) * 16, 0, 0);
        v16h bf = *(const v16h*)bptr;
        acc = __builtin_amdgcn_wmma_f32_16x16x32_f16(false, af, false, bf,
                                                     (short)0, acc, false, false);
    }
    int col   = wave * 16 + (lane & 15);
    int rbase = (lane >> 4) * 8;
    float bias = b_lin[col];
    for (int v = 0; v < 8; ++v) {
        int row = nBase + rbase + v;
        xout[(size_t)row * Ff + col] = acc[v] + bias;
    }
}

// ---------------------------------------------------------------------------
// Classifier: logits[b][c][i][j] = sum_f relu(xout[b, ho=j, wo=i, f])*Wcl[c][f]+bcl[c]
// flat logits layout [B][NC][Wo][Ho] (matches reference flat order).
// ---------------------------------------------------------------------------
__global__ void classifier_kernel(const float* __restrict__ xout,
                                  const float* __restrict__ Wcl,
                                  const float* __restrict__ bcl,
                                  float* __restrict__ logits) {
    int t = blockIdx.x * 256 + threadIdx.x;
    if (t >= Ntot) return;
    int b = t / P, s = t % P;
    int i = s / Ho, j = s % Ho;   // i: Wo index, j: Ho index
    const float* row = xout + (size_t)(b * P + j * Wo + i) * Ff;
    float a[NC];
    for (int c = 0; c < NC; ++c) a[c] = bcl[c];
    for (int f = 0; f < Ff; ++f) {
        float rv = fmaxf(row[f], 0.0f);
        for (int c = 0; c < NC; ++c) a[c] += rv * Wcl[c * Ff + f];
    }
    float* lp = logits + (size_t)b * (NC * P) + s;
    for (int c = 0; c < NC; ++c) lp[(size_t)c * P] = a[c];
}

// Per-batch softmax stats (max, sum of exp) over NC*P values, one block/batch.
__global__ void softmax_stats_kernel(const float* __restrict__ logits,
                                     float* __restrict__ redbuf) {
    __shared__ float red[256];
    int b = blockIdx.x, tid = threadIdx.x;
    const float* lp = logits + (size_t)b * (NC * P);
    float mx = -3.0e38f;
    for (int i = tid; i < NC * P; i += 256) mx = fmaxf(mx, lp[i]);
    red[tid] = mx; __syncthreads();
    for (int o = 128; o > 0; o >>= 1) {
        if (tid < o) red[tid] = fmaxf(red[tid], red[tid + o]);
        __syncthreads();
    }
    float m = red[0]; __syncthreads();
    float sm = 0.0f;
    for (int i = tid; i < NC * P; i += 256) sm += expf(lp[i] - m);
    red[tid] = sm; __syncthreads();
    for (int o = 128; o > 0; o >>= 1) {
        if (tid < o) red[tid] += red[tid + o];
        __syncthreads();
    }
    if (tid == 0) { redbuf[b] = m; redbuf[8 + b] = red[0]; }
}

// probs -> d_out (OFF_PROBS) and likelihood[b][c] -> d_out (OFF_LIKE).
__global__ void probs_like_kernel(const float* __restrict__ logits,
                                  const float* __restrict__ redbuf,
                                  float* __restrict__ dout) {
    __shared__ float red[256];
    int bc = blockIdx.x, tid = threadIdx.x;
    int b = bc / NC;
    float m = redbuf[b];
    float inv = 1.0f / redbuf[8 + b];
    const float* lp = logits + (size_t)b * (NC * P) + (size_t)(bc % NC) * P;
    float* pp = dout + OFF_PROBS + (size_t)bc * P;
    float acc = 0.0f;
    for (int s = tid; s < P; s += 256) {
        float p = expf(lp[s] - m) * inv;
        pp[s] = p;
        acc += p;
    }
    red[tid] = acc; __syncthreads();
    for (int o = 128; o > 0; o >>= 1) {
        if (tid < o) red[tid] += red[tid + o];
        __syncthreads();
    }
    if (tid == 0) dout[OFF_LIKE + bc] = red[0];
}

// Detection + boxes, one block per batch.
__global__ void detect_kernel(const float* __restrict__ out6,
                              const int* __restrict__ image_dim,
                              float* __restrict__ dout) {
    __shared__ float rv[256];
    __shared__ int   ri[256];
    int b = blockIdx.x, tid = threadIdx.x;
    const float* like = dout + OFF_LIKE + b * NC;
    int pred = 0; float bvl = like[0];
    for (int c = 1; c < NC; ++c) { float v = like[c]; if (v > bvl) { bvl = v; pred = c; } }
    const float* sc = dout + OFF_PROBS + (size_t)(b * NC + pred) * P;
    float best = -3.0e38f; int bi = 0x7fffffff;
    for (int s = tid; s < P; s += 256) {
        float v = sc[s];
        if (v > best || (v == best && s < bi)) { best = v; bi = s; }
    }
    rv[tid] = best; ri[tid] = bi; __syncthreads();
    for (int o = 128; o > 0; o >>= 1) {
        if (tid < o) {
            if (rv[tid + o] > rv[tid] ||
                (rv[tid + o] == rv[tid] && ri[tid + o] < ri[tid])) {
                rv[tid] = rv[tid + o]; ri[tid] = ri[tid + o];
            }
        }
        __syncthreads();
    }
    if (tid == 0) {
        int det = ri[0];
        float conf = rv[0];
        int loc_y = det / Wo, loc_x = det % Wo;
        int ho = loc_x, wo = loc_y;   // det2 = loc_x*Ho + loc_y -> pixel (ho,wo)
        float th[6];
        for (int j = 0; j < 6; ++j)
            th[j] = out6[((size_t)(b * 6 + j) * Ho + wo) * Wo + ho];
        float dim = (float)image_dim[0];
        float mnx = 3e38f, mny = 3e38f, mxx = -3e38f, mxy = -3e38f;
        float mnxn = 3e38f, mnyn = 3e38f, mxxn = -3e38f, mxyn = -3e38f;
        for (int kh = 0; kh < 3; ++kh)
            for (int kw = 0; kw < 3; ++kw) {
                float cxv = (float)kw - 1.0f, cyv = (float)kh - 1.0f;
                float gx = th[0] * cxv + th[1] * cyv + th[2];
                float gy = th[3] * cxv + th[4] * cyv + th[5];
                float tx = (gx * 1.5f + (float)ho + 1.5f) * (1.0f / 95.0f);
                float ty = (gy * 1.5f + (float)wo + 1.5f) * (1.0f / 95.0f);
                mnx = fminf(mnx, tx); mxx = fmaxf(mxx, tx);
                mny = fminf(mny, ty); mxy = fmaxf(mxy, ty);
                float txn = (cxv * 1.5f + (float)ho + 1.5f) * (1.0f / 95.0f);
                float tyn = (cyv * 1.5f + (float)wo + 1.5f) * (1.0f / 95.0f);
                mnxn = fminf(mnxn, txn); mxxn = fmaxf(mxxn, txn);
                mnyn = fminf(mnyn, tyn); mxyn = fmaxf(mxyn, tyn);
            }
        float* box = dout + OFF_BOX + b * 5;
        box[0] = fminf(fmaxf(mnx * dim, 0.0f), dim);
        box[1] = fminf(fmaxf(mny * dim, 0.0f), dim);
        box[2] = fminf(fmaxf(mxx * dim, 0.0f), dim);
        box[3] = fminf(fmaxf(mxy * dim, 0.0f), dim);
        box[4] = conf;
        float* boxnt = dout + OFF_BOXNT + b * 5;
        boxnt[0] = fminf(fmaxf(mnxn * dim, 0.0f), dim);
        boxnt[1] = fminf(fmaxf(mnyn * dim, 0.0f), dim);
        boxnt[2] = fminf(fmaxf(mxxn * dim, 0.0f), dim);
        boxnt[3] = fminf(fmaxf(mxyn * dim, 0.0f), dim);
        boxnt[4] = conf;
    }
}

extern "C" void kernel_launch(void* const* d_in, const int* in_sizes, int n_in,
                              void* d_out, int out_size, void* d_ws, size_t ws_size,
                              hipStream_t stream) {
    const float* x       = (const float*)d_in[0];
    const int*   img_dim = (const int*)  d_in[1];
    const float* stn0_w  = (const float*)d_in[2];
    const float* stn0_b  = (const float*)d_in[3];
    const float* stn1_w  = (const float*)d_in[4];
    const float* W_lin   = (const float*)d_in[5];
    const float* b_lin   = (const float*)d_in[6];
    const float* W_cl    = (const float*)d_in[7];
    const float* b_cl    = (const float*)d_in[8];
    float* dout = (float*)d_out;

    // workspace carving (all 32B aligned)
    float* h      = (float*)d_ws;                    // 4718592
    float* out6   = h + (size_t)Bsz * CH * HW;       // 424128
    float* xout   = out6 + (size_t)Bsz * 6 * P;      // 9048064
    float* logits = xout + (size_t)Ntot * Ff;        // 1413760
    float* redbuf = logits + (size_t)Bsz * NC * P;   // 16
    _Float16* bp0 = (_Float16*)(redbuf + 16);        // 36864 halves
    _Float16* bpl = bp0 + (size_t)KB * 4 * 32 * 16;  // 73728 halves

    pack_b_kernel<<<(KB * 4 * 512 + 255) / 256, 256, 0, stream>>>(stn0_w, bp0, 4);
    pack_b_kernel<<<(KB * 8 * 512 + 255) / 256, 256, 0, stream>>>(W_lin, bpl, 8);

    conv0_wmma_kernel<<<(Bsz * HW) / 16, 128, 0, stream>>>(x, bp0, stn0_b, h);
    conv1_theta_kernel<<<(Ntot + 255) / 256, 256, 0, stream>>>(
        h, stn1_w, out6, dout + OFF_THETA);
    gather_linear_wmma_kernel<<<Ntot / 16, 256, 0, stream>>>(
        x, out6, bpl, b_lin, xout);
    classifier_kernel<<<(Ntot + 255) / 256, 256, 0, stream>>>(
        xout, W_cl, b_cl, logits);
    softmax_stats_kernel<<<Bsz, 256, 0, stream>>>(logits, redbuf);
    probs_like_kernel<<<Bsz * NC, 256, 0, stream>>>(logits, redbuf, dout);
    detect_kernel<<<Bsz, 256, 0, stream>>>(out6, img_dim, dout);
}
